// CrossAttention_23862838296718
// MI455X (gfx1250) — matrix operand order repaired and checked by reading, not compiled, not run on hardware
//
#include <hip/hip_runtime.h>
#include <hip/hip_bf16.h>
#include <cmath>

// ---------------------------------------------------------------------------
// CDNA5 (gfx1250) cross-attention.
//   q@Wq, ctx@Wk, ctx@Wv        : bf16 WMMA GEMMs (f32 accum)
//   S = QK^T + softmax          : fused in 128KB LDS (320KB/WGP), TDM-staged
//                                 Q tile + async-LDS double-buffered K tiles
//   O = P@V                     : bf16 WMMA, heads merged on store
// Wave32; v_wmma_f32_16x16x32_bf16 everywhere; tensor_load_to_lds +
// global_load_async_to_lds_b128 + s_wait_tensorcnt/asynccnt data movement.
// ---------------------------------------------------------------------------

typedef __attribute__((ext_vector_type(16))) __bf16        v16bf;
typedef __attribute__((ext_vector_type(8)))  float         v8f;
typedef __attribute__((ext_vector_type(8)))  unsigned int  v8u;
typedef __attribute__((ext_vector_type(4)))  unsigned int  v4u;
typedef __attribute__((ext_vector_type(8)))  int           v8i;
typedef __attribute__((ext_vector_type(4)))  int           v4i;

#define DEV static __device__ __forceinline__

// ---- LDS layout for the fused scores kernel (bytes) -----------------------
#define SCORES_F       (16 * 2048)                    // f32 score tile
#define QTILE_OFF      ((SCORES_F + 256) * 4)         // 132096
#define QROW_BYTES     144                            // 64 bf16 + 16B pad
#define KBUF_OFF       (QTILE_OFF + 16 * QROW_BYTES)  // 134400
#define KBUF_BUF       (16 * QROW_BYTES)              // 2304 per buffer
#define KBUF_WAVE      (2 * KBUF_BUF)                 // double buffered
#define SMEM_BYTES     (KBUF_OFF + 8 * KBUF_WAVE)     // 171264 < 320KB/WGP
#define KTILE_GSTRIDE  (16ull * 512ull * 2ull)        // 16KB per K n-tile

// ---------------------------------------------------------------------------
// f32 -> bf16 (RNE), packed pair.  Prefer the HW packed converter.
// ---------------------------------------------------------------------------
DEV unsigned int pk2bf(float a, float b) {
#if __has_builtin(__builtin_amdgcn_cvt_pk_bf16_f32)
  typedef __attribute__((ext_vector_type(2))) __bf16 v2bf;
  v2bf r = __builtin_amdgcn_cvt_pk_bf16_f32(a, b);
  return __builtin_bit_cast(unsigned int, r);
#else
  unsigned int ua = __float_as_uint(a), ub = __float_as_uint(b);
  ua = (ua + 0x7FFFu + ((ua >> 16) & 1u)) >> 16;
  ub = (ub + 0x7FFFu + ((ub >> 16) & 1u)) >> 16;
  return (ua & 0xFFFFu) | (ub << 16);
#endif
}

DEV unsigned short f2bf(float f) {
  return (unsigned short)(pk2bf(f, 0.f) & 0xFFFFu);
}

DEV v16bf make_v16bf(uint4 x, uint4 y) {
  v8u r;
  r[0] = x.x; r[1] = x.y; r[2] = x.z; r[3] = x.w;
  r[4] = y.x; r[5] = y.y; r[6] = y.z; r[7] = y.w;
  return __builtin_bit_cast(v16bf, r);
}

// A fragment (16x32 bf16): lanes 0-15 hold K=0..7,16..23; lanes16-31 8..15,24..31.
DEV v16bf load_a_bf16(const unsigned short* row, int lane) {
  int kb = (lane >> 4) << 3;
  uint4 g0 = *(const uint4*)(row + kb);
  uint4 g1 = *(const uint4*)(row + kb + 16);
  return make_v16bf(g0, g1);
}

// A fragment from f32 source, converted on the fly.
DEV v16bf load_a_f32(const float* row, int lane) {
  int kb = (lane >> 4) << 3;
  float4 f0 = *(const float4*)(row + kb);
  float4 f1 = *(const float4*)(row + kb + 4);
  float4 f2 = *(const float4*)(row + kb + 16);
  float4 f3 = *(const float4*)(row + kb + 20);
  v8u r;
  r[0] = pk2bf(f0.x, f0.y); r[1] = pk2bf(f0.z, f0.w);
  r[2] = pk2bf(f1.x, f1.y); r[3] = pk2bf(f1.z, f1.w);
  r[4] = pk2bf(f2.x, f2.y); r[5] = pk2bf(f2.z, f2.w);
  r[6] = pk2bf(f3.x, f3.y); r[7] = pk2bf(f3.z, f3.w);
  return __builtin_bit_cast(v16bf, r);
}

// B fragment (32x16 bf16): lane = column n; lanes 0-15 hold K=0..7,16..23; lanes16-31 8..15,24..31.
DEV v16bf load_b_bf16(const unsigned short* row, int lane) {
  int kb = (lane >> 4) << 3;
  uint4 g0 = *(const uint4*)(row + kb);
  uint4 g1 = *(const uint4*)(row + kb + 16);
  return make_v16bf(g0, g1);
}

DEV v8f wmma_bf16(v16bf a, v16bf b, v8f c) {
  return __builtin_amdgcn_wmma_f32_16x16x32_bf16(false, a, false, b,
                                                 (short)0, c, false, false);
}

// ---- CDNA5 sync / async helpers -------------------------------------------
DEV void wait_async0() {
#if __has_builtin(__builtin_amdgcn_s_wait_asynccnt)
  __builtin_amdgcn_s_wait_asynccnt(0);
#else
  asm volatile("s_wait_asynccnt 0x0" ::: "memory");
#endif
}
DEV void wait_async4() {
#if __has_builtin(__builtin_amdgcn_s_wait_asynccnt)
  __builtin_amdgcn_s_wait_asynccnt(4);
#else
  asm volatile("s_wait_asynccnt 0x4" ::: "memory");
#endif
}

// One per-lane 16B global->LDS async copy (GLOBAL_LOAD_ASYNC_TO_LDS_B128).
DEV void async_g2l_b128(unsigned lds_addr, unsigned long long gaddr) {
  asm volatile("global_load_async_to_lds_b128 %0, %1, off"
               :: "v"(lds_addr), "v"(gaddr)
               : "memory");
}

// TDM: load a 16x64 bf16 tile (row stride 512 elems) into LDS with 16B row
// padding (144B pitch).  D# per CDNA5 ISA ch.8.
DEV void tdm_load_qtile_16x64(const unsigned short* gsrc, unsigned lds_addr) {
#if __has_builtin(__builtin_amdgcn_tensor_load_to_lds)
  unsigned long long ga = (unsigned long long)(size_t)gsrc;
  v4u g0;
  g0[0] = 1u;                                          // count=1 (valid D#)
  g0[1] = lds_addr;                                    // lds_addr
  g0[2] = (unsigned)(ga & 0xFFFFFFFFu);                // global_addr[31:0]
  g0[3] = (unsigned)((ga >> 32) & 0x01FFFFFFu) | (2u << 30);  // [56:32] | type=2
  v8i g1;
  //   data_size=1 (2B) @16 | pad_enable @20 | pad_interval=4 (32dw) @22
  // | pad_amount=3 (4dw=16B) @25   ; workgroup_mask=0 (not in cluster)
  g1[0] = (int)((1u << 16) | (1u << 20) | (4u << 22) | (3u << 25));
  g1[1] = (int)(64u << 16);     // tensor_dim0[15:0]=64
  g1[2] = (int)(16u << 16);     // tensor_dim0[31:16]=0 | tensor_dim1[15:0]=16
  g1[3] = (int)(64u << 16);     // tensor_dim1 hi=0 | tile_dim0=64
  g1[4] = (int)16u;             // tile_dim1=16 | tile_dim2=0
  g1[5] = (int)512;             // tensor_dim0_stride[31:0]=512
  g1[6] = 0;                    // stride hi | dim1_stride lo
  g1[7] = 0;
  v4i z4 = {0, 0, 0, 0};
#if defined(__clang_major__) && (__clang_major__ >= 23)
  v8i z8 = {0, 0, 0, 0, 0, 0, 0, 0};
  __builtin_amdgcn_tensor_load_to_lds(g0, g1, z4, z4, z8, 0);
#else
  __builtin_amdgcn_tensor_load_to_lds(g0, g1, z4, z4, 0);
#endif
#else
  (void)gsrc; (void)lds_addr;   // fallback handled by caller
#endif
}

DEV void wait_tensor0() {
#if __has_builtin(__builtin_amdgcn_s_wait_tensorcnt)
  __builtin_amdgcn_s_wait_tensorcnt((short)0);
#else
  asm volatile("s_wait_tensorcnt 0x0" ::: "memory");
#endif
}

// ---------------------------------------------------------------------------
// Kernel 1: W[k][n] f32 -> WT[n][k] bf16 (so WMMA B fragments are contiguous).
// ---------------------------------------------------------------------------
__global__ void prep_wT(const float* __restrict__ W, unsigned short* __restrict__ WT,
                        int K, int N) {
  int idx = blockIdx.x * blockDim.x + threadIdx.x;
  if (idx >= K * N) return;
  int n = idx / K, k = idx - n * K;
  WT[idx] = f2bf(W[(size_t)k * N + n]);
}

// ---------------------------------------------------------------------------
// Kernel 2: projection GEMM. C[m,n] = scale * sum_k X[m,k]*WT[n,k], bf16 out.
// MODE 0: out[m*N + n]          (Q-head, K-head natural layout)
// MODE 1: V^T per head:  out[((b*8+h)*64+d)*2048 + nk]  (packed 16B stores)
// ---------------------------------------------------------------------------
template <int MODE>
__global__ void gemm_proj(const float* __restrict__ X,
                          const unsigned short* __restrict__ WT,
                          unsigned short* __restrict__ out,
                          int M, int K, int N, float scale) {
  int wave = threadIdx.x >> 5, lane = threadIdx.x & 31;
  int t = blockIdx.x * 8 + wave;
  int tilesN = N >> 4;
  int tm = t / tilesN, tn = t - tm * tilesN;
  const float*          arow = X  + (size_t)(tm * 16 + (lane & 15)) * K;
  const unsigned short* brow = WT + (size_t)(tn * 16 + (lane & 15)) * K;
  v8f c = {0.f, 0.f, 0.f, 0.f, 0.f, 0.f, 0.f, 0.f};
  for (int kc = 0; kc < K; kc += 32, arow += 32, brow += 32) {
    __builtin_prefetch(arow + 64, 0, 1);
    v16bf a = load_a_f32(arow, lane);
    v16bf b = load_b_bf16(brow, lane);
    c = wmma_bf16(a, b, c);
  }
  int mhi = (lane >> 4) << 3;
  int n   = tn * 16 + (lane & 15);
  int m0  = tm * 16 + mhi;
  if (MODE == 0) {
    unsigned short* o = out + (size_t)m0 * N + n;
#pragma unroll
    for (int v = 0; v < 8; ++v) o[(size_t)v * N] = f2bf(c[v] * scale);
  } else {
    int b8 = m0 >> 11, nk0 = m0 & 2047;          // M = B*2048
    int hh = n >> 6,   d   = n & 63;             // N = 512 = 8 heads * 64
    uint4 pk;
    pk.x = pk2bf(c[0] * scale, c[1] * scale);
    pk.y = pk2bf(c[2] * scale, c[3] * scale);
    pk.z = pk2bf(c[4] * scale, c[5] * scale);
    pk.w = pk2bf(c[6] * scale, c[7] * scale);
    *(uint4*)(out + ((size_t)((b8 * 8 + hh) * 64 + d)) * 2048 + nk0) = pk;
  }
}

// ---------------------------------------------------------------------------
// Kernel 3: fused scores + softmax.  One block = 16 query rows of one (b,h).
//  - Q tile (16x64 bf16) staged by the Tensor Data Mover (padded 144B rows).
//  - K tiles (16x64 bf16) double-buffered per wave via async global->LDS;
//    per-lane global addresses kept in registers and bumped by +16KB/tile.
//  - S tile (16x2048 f32) lives entirely in LDS; softmax never touches HBM
//    until the single normalized write.
// ---------------------------------------------------------------------------
__global__ void attn_scores_softmax(const unsigned short* __restrict__ qh,
                                    const unsigned short* __restrict__ kh,
                                    float* __restrict__ attn) {
  extern __shared__ char smem_raw[];
  float* smem = (float*)smem_raw;                  // 16x2048 scores
  float* red  = smem + SCORES_F;                   // 256 reduction slots
  unsigned short* qtile = (unsigned short*)(smem_raw + QTILE_OFF);
  char* kbuf = smem_raw + KBUF_OFF;

  int wave = threadIdx.x >> 5, lane = threadIdx.x & 31;
  int qt = blockIdx.x & 63;
  int h  = (blockIdx.x >> 6) & 7;
  int b  = blockIdx.x >> 9;

  // ---- stage Q tile into LDS ------------------------------------------------
  const unsigned short* qsrc = qh + ((size_t)(b * 1024 + qt * 16)) * 512 + h * 64;
#if __has_builtin(__builtin_amdgcn_tensor_load_to_lds)
  if (wave == 0)
    tdm_load_qtile_16x64(qsrc, (unsigned)(size_t)qtile);
  wait_tensor0();
#else
  if (threadIdx.x < 128) {                     // 16 rows x 8 chunks of 16B
    int r = threadIdx.x >> 3, ch = threadIdx.x & 7;
    async_g2l_b128((unsigned)(size_t)(smem_raw + QTILE_OFF) + r * QROW_BYTES + ch * 16,
                   (unsigned long long)(size_t)((const char*)(qsrc + (size_t)r * 512) + ch * 16));
  }
  wait_async0();
#endif
  __syncthreads();

  int m   = lane & 15;
  int mhi = (lane >> 4) << 3;
  const unsigned short* qrow = qtile + m * (QROW_BYTES / 2);
  v16bf a0 = load_a_bf16(qrow, lane);          // d = 0..31
  v16bf a1 = load_a_bf16(qrow + 32, lane);     // d = 32..63

  // ---- K tiles: double-buffered async staging + WMMA ------------------------
  char* kb0 = kbuf + wave * KBUF_WAVE;
  const unsigned short* khb = kh + ((size_t)(b * 2048)) * 512 + h * 64;
  unsigned kb0_lds = (unsigned)(size_t)kb0;

  // Per-lane source addresses (4 x 16B per tile), bumped by +16KB per n-tile;
  // LDS destinations precomputed for both buffers.
  unsigned            lds0[4], lds1[4];
  unsigned long long  ga[4];
#pragma unroll
  for (int j = 0; j < 4; ++j) {
    int lin = j * 32 + lane, r = lin >> 3, ch = lin & 7;
    lds0[j] = kb0_lds + r * QROW_BYTES + ch * 16;
    lds1[j] = lds0[j] + KBUF_BUF;
    ga[j]   = (unsigned long long)(size_t)(
                  (const char*)(khb + (size_t)(wave * 256 + r) * 512) + ch * 16);
  }
#pragma unroll
  for (int j = 0; j < 4; ++j) { async_g2l_b128(lds0[j], ga[j]); ga[j] += KTILE_GSTRIDE; }

  const unsigned short* krow0 = (const unsigned short*)kb0 + m * (QROW_BYTES / 2);
  const unsigned short* krow1 = krow0 + KBUF_BUF / 2;
  int ncol = wave * 256 + m;

  for (int i = 0; i < 16; ++i, ncol += 16) {
    if (i < 15) {                                  // prefetch next buffer
#pragma unroll
      for (int j = 0; j < 4; ++j) {
        async_g2l_b128(((i + 1) & 1) ? lds1[j] : lds0[j], ga[j]);
        ga[j] += KTILE_GSTRIDE;
      }
      wait_async4();                               // current buffer complete
    } else {
      wait_async0();
    }
    const unsigned short* krow = (i & 1) ? krow1 : krow0;
    v16bf b0 = load_b_bf16(krow, lane);
    v16bf b1 = load_b_bf16(krow + 32, lane);
    v8f c = {0.f, 0.f, 0.f, 0.f, 0.f, 0.f, 0.f, 0.f};
    c = wmma_bf16(a0, b0, c);
    c = wmma_bf16(a1, b1, c);
#pragma unroll
    for (int v = 0; v < 8; ++v) smem[(size_t)(mhi + v) * 2048 + ncol] = c[v];
  }
  __syncthreads();

  // ---- softmax: 16 threads per row, 128 contiguous cols each ----------------
  int r = threadIdx.x >> 4, ci = threadIdx.x & 15;
  float* rowp = smem + (size_t)r * 2048 + ci * 128;

  float mx = -3.402823466e38f;
  for (int i = 0; i < 128; ++i) mx = fmaxf(mx, rowp[i]);
  red[r * 16 + ci] = mx;
  __syncthreads();
  float rmax = -3.402823466e38f;
  for (int i = 0; i < 16; ++i) rmax = fmaxf(rmax, red[r * 16 + i]);
  __syncthreads();                               // red reused below

  float s = 0.f;
  for (int i = 0; i < 128; ++i) {
    float e = __expf(rowp[i] - rmax);
    rowp[i] = e;
    s += e;
  }
  red[r * 16 + ci] = s;
  __syncthreads();
  float rsum = 0.f;
  for (int i = 0; i < 16; ++i) rsum += red[r * 16 + i];
  float inv = 1.0f / rsum;

  float4* dst = (float4*)(attn +
      ((size_t)((b * 8 + h) * 1024 + qt * 16 + r)) * 2048 + ci * 128);
  float4* src = (float4*)rowp;
  for (int i = 0; i < 32; ++i) {
    float4 v = src[i];
    v.x *= inv; v.y *= inv; v.z *= inv; v.w *= inv;
    dst[i] = v;
  }
}

// ---------------------------------------------------------------------------
// Kernel 4: O = P @ V per (b,h).  P (f32) -> bf16 per A fragment; V^T rows
// contiguous.  Heads merged on store: out[b][q][h*64+d].
// ---------------------------------------------------------------------------
__global__ void attn_pv(const float* __restrict__ attn,
                        const unsigned short* __restrict__ vt,
                        float* __restrict__ outc) {
  int wave = threadIdx.x >> 5, lane = threadIdx.x & 31;
  int qt2 = blockIdx.x & 31;
  int h   = (blockIdx.x >> 5) & 7;
  int b   = blockIdx.x >> 8;
  int qsub = wave >> 2, dt = wave & 3;

  int q = qt2 * 32 + qsub * 16 + (lane & 15);
  const float* arow = attn + ((size_t)((b * 8 + h) * 1024 + q)) * 2048;
  const unsigned short* brow =
      vt + ((size_t)((b * 8 + h) * 64 + dt * 16 + (lane & 15))) * 2048;

  v8f c = {0.f, 0.f, 0.f, 0.f, 0.f, 0.f, 0.f, 0.f};
  for (int kc = 0; kc < 2048; kc += 32, arow += 32, brow += 32) {
    __builtin_prefetch(arow + 64, 0, 1);
    v16bf a  = load_a_f32(arow, lane);
    v16bf bb = load_b_bf16(brow, lane);
    c = wmma_bf16(a, bb, c);
  }
  int mhi  = (lane >> 4) << 3;
  int ncol = h * 64 + dt * 16 + (lane & 15);
  float* o = outc + ((size_t)(b * 1024 + qt2 * 32 + qsub * 16 + mhi)) * 512 + ncol;
#pragma unroll
  for (int v = 0; v < 8; ++v) o[(size_t)v * 512] = c[v];
}

// ---------------------------------------------------------------------------
// Host launcher.
// inputs: q[4,1024,512], context[4,2048,1024], Wq[512,512], Wk/Wv[1024,512]
// d_out:  ctx_vec (4*1024*512 f32) ++ attn_weights (4*8*1024*2048 f32)
// ---------------------------------------------------------------------------
extern "C" void kernel_launch(void* const* d_in, const int* in_sizes, int n_in,
                              void* d_out, int out_size, void* d_ws, size_t ws_size,
                              hipStream_t stream) {
  const float* q   = (const float*)d_in[0];
  const float* ctx = (const float*)d_in[1];
  const float* Wq  = (const float*)d_in[2];
  const float* Wk  = (const float*)d_in[3];
  const float* Wv  = (const float*)d_in[4];

  float* out_ctx  = (float*)d_out;
  float* out_attn = out_ctx + (size_t)4 * 1024 * 512;

  char* ws = (char*)d_ws;
  unsigned short* wqT = (unsigned short*)ws; ws += (size_t)512 * 512  * 2;
  unsigned short* wkT = (unsigned short*)ws; ws += (size_t)512 * 1024 * 2;
  unsigned short* wvT = (unsigned short*)ws; ws += (size_t)512 * 1024 * 2;
  unsigned short* qh  = (unsigned short*)ws; ws += (size_t)4 * 1024 * 512 * 2;
  unsigned short* kh  = (unsigned short*)ws; ws += (size_t)4 * 2048 * 512 * 2;
  unsigned short* vt  = (unsigned short*)ws;  // 4*8*64*2048 bf16

  prep_wT<<<(512 * 512  + 255) / 256, 256, 0, stream>>>(Wq, wqT, 512,  512);
  prep_wT<<<(1024 * 512 + 255) / 256, 256, 0, stream>>>(Wk, wkT, 1024, 512);
  prep_wT<<<(1024 * 512 + 255) / 256, 256, 0, stream>>>(Wv, wvT, 1024, 512);

  const float scale = 1.0f / sqrtf(512.0f);      // folded into Q projection
  gemm_proj<0><<<1024, 256, 0, stream>>>(q,   wqT, qh, 4096, 512,  512, scale);
  gemm_proj<0><<<2048, 256, 0, stream>>>(ctx, wkT, kh, 8192, 1024, 512, 1.0f);
  gemm_proj<1><<<2048, 256, 0, stream>>>(ctx, wvT, vt, 8192, 1024, 512, 1.0f);

  attn_scores_softmax<<<4 * 8 * 64, 256, SMEM_BYTES, stream>>>(qh, kh, out_attn);
  attn_pv<<<4 * 8 * 32, 256, 0, stream>>>(out_attn, vt, out_ctx);
}
